// TopKBalancedNoisyGate_15307263443371
// MI455X (gfx1250) — matrix operand order repaired and checked by reading, not compiled, not run on hardware
//
#include <hip/hip_runtime.h>
#include <hip/hip_bf16.h>
#include <math.h>

typedef __attribute__((ext_vector_type(16))) __bf16 v16bf;
typedef __attribute__((ext_vector_type(8)))  float  v8f;

#define N_TOK 16384
#define DIM   4096
#define NE    64
#define NEP   65   // padded f32 LDS row stride (bank-conflict-free row scans)
#define KSEL  8

#define WAVES 4
#define TPB   (WAVES * 32)
#define KB_TOTAL (DIM / 32)           // 128 k-steps of 32
#define KB_PER_WAVE (KB_TOTAL / WAVES) // 32 per wave (K split across waves)
#define NBLOCKS (N_TOK / 16)          // one 16-row tile per block -> 1024 blocks

// d_out flat layout (reference tuple order):
// top_k_indices[16384*8], top_k_scores[16384*8], balance_loss[1], load[64], importance[64]
#define OFF_IDX  0
#define OFF_SC   (N_TOK * KSEL)
#define OFF_BAL  (2 * N_TOK * KSEL)
#define OFF_LOAD (OFF_BAL + 1)
#define OFF_IMP  (OFF_LOAD + NE)
#define OUT_TOTAL (OFF_IMP + NE)

// ---------------------------------------------------------------------------
// Pack a (K x 64) f32 row-major weight into bf16 B-fragment order for
// V_WMMA_F32_16X16X32_BF16:
//   fragment index = ((kb*4 + t)*32 + lane)*16 + e
//   lane <  16 : element e = W[kb*32 + (e<8 ? e : 8+e)][t*16 + lane]       (K = 0..7,16..23)
//   lane >= 16 : element e = W[kb*32 + (e<8 ? 8+e : 16+e)][t*16 + lane - 16] (K = 8..15,24..31)
// Each wave then loads its B fragment as one contiguous v16bf (32 B).
// ---------------------------------------------------------------------------
__global__ void gate_pack_kernel(const float* __restrict__ W,
                                 __bf16* __restrict__ dst, int total) {
    int idx = blockIdx.x * blockDim.x + threadIdx.x;
    if (idx >= total) return;
    int e    = idx & 15;
    int lane = (idx >> 4) & 31;
    int t    = (idx >> 9) & 3;
    int kb   = idx >> 11;
    int k, n;
    if (lane < 16) { k = kb * 32 + (e < 8 ? e : 8 + e);      n = t * 16 + lane; }
    else           { k = kb * 32 + (e < 8 ? 8 + e : 16 + e); n = t * 16 + (lane - 16); }
    dst[idx] = (__bf16)W[k * NE + n];
}

__global__ void gate_init_kernel(float* __restrict__ out) {
    int i = threadIdx.x;
    if (i < (OUT_TOTAL - OFF_BAL)) out[OFF_BAL + i] = 0.0f;
}

__device__ __forceinline__ float softplus_f(float v) {
    return (v > 20.0f) ? v : log1pf(expf(v));
}

// ---------------------------------------------------------------------------
// Main fused kernel: one 16-row token tile per block; K split across 4 waves
// (4096 waves total -> ~4 waves/SIMD for latency hiding on the x stream).
//   h1 = x@Wg1, h2 = x@Wn   (bf16 WMMA, f32 accum; A fragment shared)
//   cross-wave partial reduction in LDS
//   logits_gate = tanh(h1)@Wg2 (wave w owns N-tile w)
//   nc = softplus(h2)+eps ; logits = lg + noise*nc
//   top-9 / softmax-8 / normcdf load / importance, atomically reduced.
// ---------------------------------------------------------------------------
__global__ __launch_bounds__(TPB) void gate_main_kernel(
    const float* __restrict__ x, const float* __restrict__ noise,
    const v16bf* __restrict__ pG, const v16bf* __restrict__ pN,
    const v16bf* __restrict__ pW2, float* __restrict__ out) {

    __shared__ float  sRedG[WAVES][16][NE]; // per-wave K-partial of x@Wg1
    __shared__ float  sRedN[WAVES][16][NE]; // per-wave K-partial of x@Wn
    __shared__ __bf16 sTanh[16][NE];        // tanh(h1), A-operand of 2nd GEMM
    __shared__ float  sNC [16][NEP];
    __shared__ float  sLG [16][NEP];
    __shared__ float  sLT [16][NEP];
    __shared__ float  sThr[16][2];          // {thr_if_in, thr_if_out} per row
    __shared__ float  sImp[NE];
    __shared__ float  sLoad[NE];

    const int tid  = threadIdx.x;
    const int lane = tid & 31;
    const int w    = tid >> 5;
    const int half = lane >> 4;
    const int l16  = lane & 15;
    const int rowBase = blockIdx.x * 16;

    if (tid < NE) { sImp[tid] = 0.0f; sLoad[tid] = 0.0f; }

    v8f accG[4] = {};
    v8f accN[4] = {};

    const float* xrow = x + (size_t)(rowBase + l16) * DIM;
    const int kb0 = w * KB_PER_WAVE;

    for (int kb = kb0; kb < kb0 + KB_PER_WAVE; ++kb) {
        // A fragment (16x32 bf16): lane<16 holds row l16, K{0..7,16..23};
        // lane>=16 holds K{8..15,24..31} — two contiguous float4-pairs each.
        const float* p0 = xrow + kb * 32 + half * 8;
        __builtin_prefetch(p0 + 8 * 32, 0, 0); // global_prefetch_b8, 1KB ahead
        float4 f0 = *(const float4*)(p0);
        float4 f1 = *(const float4*)(p0 + 4);
        float4 f2 = *(const float4*)(p0 + 16);
        float4 f3 = *(const float4*)(p0 + 20);
        v16bf a;
        a[0]  = (__bf16)f0.x; a[1]  = (__bf16)f0.y; a[2]  = (__bf16)f0.z; a[3]  = (__bf16)f0.w;
        a[4]  = (__bf16)f1.x; a[5]  = (__bf16)f1.y; a[6]  = (__bf16)f1.z; a[7]  = (__bf16)f1.w;
        a[8]  = (__bf16)f2.x; a[9]  = (__bf16)f2.y; a[10] = (__bf16)f2.z; a[11] = (__bf16)f2.w;
        a[12] = (__bf16)f3.x; a[13] = (__bf16)f3.y; a[14] = (__bf16)f3.z; a[15] = (__bf16)f3.w;

        const int bi = kb * 4 * 32 + lane;
#pragma unroll
        for (int t = 0; t < 4; ++t) {
            v16bf bG = pG[bi + t * 32];
            v16bf bN = pN[bi + t * 32];
            accG[t] = __builtin_amdgcn_wmma_f32_16x16x32_bf16(
                false, a, false, bG, (short)0, accG[t], false, false);
            accN[t] = __builtin_amdgcn_wmma_f32_16x16x32_bf16(
                false, a, false, bN, (short)0, accN[t], false, false);
        }
    }

    // Stash K-partials. C layout: VGPR r, lanes 0-15 -> (M=r, N=lane);
    // lanes 16-31 -> (M=8+r, N=lane-16).
#pragma unroll
    for (int t = 0; t < 4; ++t) {
#pragma unroll
        for (int r = 0; r < 8; ++r) {
            int M = r + half * 8;
            int col = t * 16 + l16;
            sRedG[w][M][col] = accG[t][r];
            sRedN[w][M][col] = accN[t][r];
        }
    }
    __syncthreads();

    // Cross-wave reduction + activations (all 128 threads, 8 elems each).
    for (int i = tid; i < 16 * NE; i += TPB) {
        int M = i >> 6, c = i & 63;
        float g = sRedG[0][M][c] + sRedG[1][M][c] + sRedG[2][M][c] + sRedG[3][M][c];
        float n = sRedN[0][M][c] + sRedN[1][M][c] + sRedN[2][M][c] + sRedN[3][M][c];
        sTanh[M][c] = (__bf16)tanhf(g);
        sNC[M][c]   = softplus_f(n) + 0.01f;
    }
    __syncthreads();

    // logits_gate = tanh(h1) @ Wg2 : wave w computes N-tile w (K=64 -> 2 steps).
    v8f accL = {};
#pragma unroll
    for (int kb2 = 0; kb2 < 2; ++kb2) {
        const __bf16* tp = &sTanh[l16][kb2 * 32 + half * 8];
        v16bf a2;
#pragma unroll
        for (int j = 0; j < 8; ++j) { a2[j] = tp[j]; a2[8 + j] = tp[16 + j]; }
        v16bf b2 = pW2[(kb2 * 4 + w) * 32 + lane];
        accL = __builtin_amdgcn_wmma_f32_16x16x32_bf16(
            false, a2, false, b2, (short)0, accL, false, false);
    }

    // Assemble logits = lg + noise * nc for this wave's tile.
#pragma unroll
    for (int r = 0; r < 8; ++r) {
        int M = r + half * 8;
        int col = w * 16 + l16;
        int grow = rowBase + M;
        float lg = accL[r];
        float nc = sNC[M][col];
        float ns = noise[(size_t)grow * NE + col];
        sLG[M][col] = lg;
        sLT[M][col] = lg + ns * nc;
    }
    __syncthreads();

    // Top-9 / softmax-8 per row (threads 0..15, fully unrolled register sort).
    if (tid < 16) {
        const int m = tid;
        const int grow = rowBase + m;
        float tv[9]; int ti9[9];
#pragma unroll
        for (int j = 0; j < 9; ++j) { tv[j] = -INFINITY; ti9[j] = 0; }
        for (int e = 0; e < NE; ++e) {
            float v = sLT[m][e];
            int ce = e;
#pragma unroll
            for (int j = 0; j < 9; ++j) { // bubble insert; ties keep lower index
                bool g = v > tv[j];
                float ov = tv[j]; int oi = ti9[j];
                tv[j] = g ? v : ov;  ti9[j] = g ? ce : oi;
                v     = g ? ov : v;  ce     = g ? oi : ce;
            }
        }
        float mx = tv[0];
        float ex[8]; float s = 0.0f;
#pragma unroll
        for (int j = 0; j < 8; ++j) { ex[j] = expf(tv[j] - mx); s += ex[j]; }
        float inv = 1.0f / s;
#pragma unroll
        for (int j = 0; j < 8; ++j) {
            float sc = ex[j] * inv;
            out[OFF_IDX + (size_t)grow * KSEL + j] = (float)ti9[j];
            out[OFF_SC  + (size_t)grow * KSEL + j] = sc;
            atomicAdd(&sImp[ti9[j]], sc);
        }
        sThr[m][0] = tv[8]; // threshold_if_in  (9th largest)
        sThr[m][1] = tv[7]; // threshold_if_out (8th largest)
    }
    __syncthreads();

    // load: Phi((lg - thr)/nc), thr selected by is_in = logits_noise > thrIn.
    // All 128 threads, 8 (row, expert) pairs each.
    for (int i = tid; i < 16 * NE; i += TPB) {
        int m = i >> 6, e = i & 63;
        float lg = sLG[m][e];
        float lt = sLT[m][e];
        float nc = sNC[m][e];
        float ln = lt - lg;
        float thr = (ln > sThr[m][0]) ? sThr[m][0] : sThr[m][1];
        float z = (lg - thr) / nc;
        float p = 0.5f * erfcf(-z * 0.70710678118654752f);
        atomicAdd(&sLoad[e], p);
    }
    __syncthreads();

    if (tid < NE) {
        atomicAdd(&out[OFF_LOAD + tid], sLoad[tid]);
        atomicAdd(&out[OFF_IMP  + tid], sImp[tid]);
    }
}

__global__ void gate_finalize_kernel(float* __restrict__ out) {
    if (threadIdx.x == 0 && blockIdx.x == 0) {
        float bal = 0.0f;
        for (int which = 0; which < 2; ++which) {
            const float* v = out + (which ? OFF_IMP : OFF_LOAD);
            float s = 0.0f;
            for (int i = 0; i < NE; ++i) s += v[i];
            float mean = s / (float)NE;
            float ss = 0.0f;
            for (int i = 0; i < NE; ++i) { float d = v[i] - mean; ss += d * d; }
            float var = ss / (float)(NE - 1); // ddof=1
            bal += var / (mean * mean + 1e-10f);
        }
        out[OFF_BAL] = 0.01f * bal;
    }
}

extern "C" void kernel_launch(void* const* d_in, const int* in_sizes, int n_in,
                              void* d_out, int out_size, void* d_ws, size_t ws_size,
                              hipStream_t stream) {
    const float* x     = (const float*)d_in[0]; // (16384, 4096)
    const float* Wg1   = (const float*)d_in[1]; // (4096, 64)
    const float* Wg2   = (const float*)d_in[2]; // (64, 64)
    const float* Wn    = (const float*)d_in[3]; // (4096, 64)
    const float* noise = (const float*)d_in[4]; // (16384, 64)
    float* out = (float*)d_out;

    __bf16* pG  = (__bf16*)d_ws;            // 4096*64 bf16 = 512 KB
    __bf16* pN  = pG + (size_t)DIM * NE;    // 512 KB
    __bf16* pW2 = pN + (size_t)DIM * NE;    // 8 KB

    gate_pack_kernel<<<(DIM * NE + 255) / 256, 256, 0, stream>>>(Wg1, pG, DIM * NE);
    gate_pack_kernel<<<(DIM * NE + 255) / 256, 256, 0, stream>>>(Wn,  pN, DIM * NE);
    gate_pack_kernel<<<(NE * NE + 255) / 256, 256, 0, stream>>>(Wg2, pW2, NE * NE);
    gate_init_kernel<<<1, 160, 0, stream>>>(out);
    gate_main_kernel<<<NBLOCKS, TPB, 0, stream>>>(
        x, noise, (const v16bf*)pG, (const v16bf*)pN, (const v16bf*)pW2, out);
    gate_finalize_kernel<<<1, 64, 0, stream>>>(out);
}
